// GraphAttentionNet_56839597195988
// MI455X (gfx1250) — compile-verified, matrix operand order checked by reading, not run on hardware
//
#include <hip/hip_runtime.h>

typedef __attribute__((ext_vector_type(16))) _Float16 v16h;
typedef __attribute__((ext_vector_type(8)))  _Float16 v8h;
typedef __attribute__((ext_vector_type(8)))  float    v8f;

#define NN   4096
#define NW   (NN / 32)      // 128 bitmask words per row
#define INF_ 512
#define HH   4
#define F1   128
#define HF1  512
#define HID1 64
#define F2   32
#define HF2  128
#define OUTF 16

// A 16-bit 16x32 fragment row (ISA 7.12.2): per lane, two contiguous 8xf16 chunks
// at K = k0 + 8*half and K = k0 + 16 + 8*half.  Same pattern serves B when B is
// stored K-contiguous per output column (i.e. transposed to [N][K]).
__device__ __forceinline__ v16h load_frag(const _Float16* base /* at K = k0 + 8*half */) {
    v8h lo = *(const v8h*)(base);
    v8h hi = *(const v8h*)(base + 16);
    return __builtin_shufflevector(lo, hi, 0, 1, 2, 3, 4, 5, 6, 7, 8, 9, 10, 11, 12, 13, 14, 15);
}

__global__ void f32_to_f16_k(const float* __restrict__ in, _Float16* __restrict__ out, int n) {
    int i = blockIdx.x * blockDim.x + threadIdx.x;
    if (i < n) out[i] = (_Float16)in[i];
}

// f32 [R,C] row-major  ->  f16 [C,R] row-major (transpose + convert), 16x16 LDS tiles.
__global__ void f32_to_f16_tr_k(const float* __restrict__ in, _Float16* __restrict__ out,
                                int R, int C) {
    __shared__ float tile[16][17];
    int tx = threadIdx.x & 15, ty = threadIdx.x >> 4;
    int r0 = blockIdx.y * 16, c0 = blockIdx.x * 16;
    int r = r0 + ty, c = c0 + tx;
    if (r < R && c < C) tile[ty][tx] = in[(size_t)r * C + c];
    __syncthreads();
    int rr = r0 + tx, cc = c0 + ty;
    if (rr < R && cc < C) out[(size_t)cc * R + rr] = (_Float16)tile[tx][ty];
}

// Pack int32 adjacency into a bitmask: 67 MB -> 2 MB (L2-resident).
__global__ void adj_to_bits_k(const int* __restrict__ adj, unsigned* __restrict__ bits) {
    int w = blockIdx.x * blockDim.x + threadIdx.x;
    if (w >= NN * NW) return;
    const int* p = adj + (size_t)w * 32;
    unsigned b = 0;
#pragma unroll
    for (int i = 0; i < 32; ++i) b |= (p[i] != 0 ? 1u : 0u) << i;
    bits[w] = b;
}

// C[M,N] = A[M,K] * BT[N,K]^T, f16 in / f32 out, one wave per 16x16 C tile.
__global__ void gemm_f16_wmma(const _Float16* __restrict__ A, const _Float16* __restrict__ BT,
                              float* __restrict__ C, int M, int N, int K) {
    int lane = threadIdx.x & 31;
    int half = lane >> 4;
    int rc   = lane & 15;
    int n0 = blockIdx.x * 16;
    int m0 = blockIdx.y * 16;
    const _Float16* arow = A  + (size_t)(m0 + rc) * K + 8 * half;
    const _Float16* brow = BT + (size_t)(n0 + rc) * K + 8 * half;
    v8f acc = {0.f, 0.f, 0.f, 0.f, 0.f, 0.f, 0.f, 0.f};
    for (int k0 = 0; k0 < K; k0 += 32) {
        v16h a = load_frag(arow + k0);
        v16h b = load_frag(brow + k0);
        acc = __builtin_amdgcn_wmma_f32_16x16x32_f16(false, a, false, b, (short)0, acc, false, false);
    }
#pragma unroll
    for (int v = 0; v < 8; ++v) {
        int m = v + 8 * half;
        C[(size_t)(m0 + m) * N + (n0 + rc)] = acc[v];
    }
}

// es[h][n] = sum_f h[n,h,f]*a_src[h,f] ; ed likewise.
__global__ void attn_scores_k(const float* __restrict__ h, const float* __restrict__ a_src,
                              const float* __restrict__ a_dst, float* __restrict__ es,
                              float* __restrict__ ed, int n, int f) {
    int idx = blockIdx.x * blockDim.x + threadIdx.x;
    if (idx >= n * HH) return;
    int node = idx / HH, head = idx % HH;
    const float* hp = h + (size_t)node * HH * f + (size_t)head * f;
    const float* as = a_src + head * f;
    const float* ad = a_dst + head * f;
    float ss = 0.f, sd = 0.f;
    for (int i = 0; i < f; ++i) { float v = hp[i]; ss += v * as[i]; sd += v * ad[i]; }
    es[head * n + node] = ss;
    ed[head * n + node] = sd;
}

// Fused masked-softmax + attn@h for one 16-row block, all 4 heads.
// hhT = per-head features transposed to [HF][NN] f16 so B fragments are contiguous.
// TPW = 16-col C tiles per wave (8 waves). F = per-head feature dim.
template <int TPW, int F>
__global__ __launch_bounds__(256) void gat_aggregate(const _Float16* __restrict__ hhT,
                                                     const unsigned* __restrict__ adjb,
                                                     const float* __restrict__ es,
                                                     const float* __restrict__ ed,
                                                     float* __restrict__ out) {
    constexpr int HF = 4 * F;
    __shared__ __align__(16) _Float16 p_lds[4 * 16 * 32];   // [head][row][j]
    __shared__ float mpart[64][4], spart[64][4];
    __shared__ float m_lds[64], s_lds[64], es_lds[64];

    int t  = threadIdx.x;
    int i0 = blockIdx.x * 16;

    // ---- Phase A: online row max + softmax denominator (4 threads per (head,row)) ----
    {
        int pair = t >> 2, sub = t & 3;          // pair = head*16 + row
        int row = pair & 15, head = pair >> 4;
        float esv = es[head * NN + i0 + row];
        if (sub == 0) es_lds[pair] = esv;
        const float*    edp = ed + (size_t)head * NN;
        const unsigned* abp = adjb + (size_t)(i0 + row) * NW;
        float m = -__builtin_inff(), s = 0.f;
        int wb = sub * (NW / 4), we = wb + NW / 4;
        for (int w = wb; w < we; ++w) {
            unsigned bits = abp[w];
            if (!bits) continue;
            int jbase = w * 32;
            for (int b = 0; b < 32; ++b) {
                if (bits & (1u << b)) {
                    float e = esv + edp[jbase + b];
                    e = (e > 0.f) ? e : 0.2f * e;     // leaky relu
                    float mn = fmaxf(m, e);
                    s = s * __expf(m - mn) + __expf(e - mn);
                    m = mn;
                }
            }
        }
        mpart[pair][sub] = m;
        spart[pair][sub] = s;
    }
    __syncthreads();
    if (t < 64) {
        float m = -__builtin_inff(), s = 0.f;
#pragma unroll
        for (int q = 0; q < 4; ++q) {
            float mq = mpart[t][q], sq = spart[t][q];
            float mn = fmaxf(m, mq);
            s = s * __expf(m - mn) + sq * __expf(mq - mn);
            m = mn;
        }
        m_lds[t] = m;
        s_lds[t] = s;
    }
    __syncthreads();

    // ---- Phase B: stream j in tiles of 32, WMMA-accumulate p @ h ----
    int wave = t >> 5, lane = t & 31;
    int half = lane >> 4, rc = lane & 15;
    int tile0  = wave * TPW;
    int head_w = (tile0 * 16) / F;                // all TPW tiles of a wave share one head
    v8f acc[TPW];
#pragma unroll
    for (int q = 0; q < TPW; ++q) acc[q] = (v8f){0.f, 0.f, 0.f, 0.f, 0.f, 0.f, 0.f, 0.f};

    for (int j0 = 0; j0 < NN; j0 += 32) {
        // cooperative probability tile: 2048 values / 256 threads
#pragma unroll
        for (int u = 0; u < 8; ++u) {
            int idx = t + u * 256;                // = head*512 + row*32 + jj
            int head = idx >> 9;
            int rem  = idx & 511;
            int row  = rem >> 5, jj = rem & 31;
            unsigned wbits = adjb[(size_t)(i0 + row) * NW + (j0 >> 5)];
            float p = 0.f;
            if ((wbits >> jj) & 1u) {
                float e = es_lds[head * 16 + row] + ed[(size_t)head * NN + j0 + jj];
                e = (e > 0.f) ? e : 0.2f * e;
                p = __expf(e - m_lds[head * 16 + row]);
            }
            p_lds[idx] = (_Float16)p;
        }
        __syncthreads();

        v16h a = load_frag(p_lds + head_w * 512 + rc * 32 + 8 * half);
#pragma unroll
        for (int q = 0; q < TPW; ++q) {
            int col = (tile0 + q) * 16 + rc;
            v16h b = load_frag(hhT + (size_t)col * NN + j0 + 8 * half);
            acc[q] = __builtin_amdgcn_wmma_f32_16x16x32_f16(false, a, false, b, (short)0,
                                                            acc[q], false, false);
        }
        __syncthreads();
    }

    // normalize by softmax denominator and store
#pragma unroll
    for (int q = 0; q < TPW; ++q) {
        int col = (tile0 + q) * 16 + rc;
#pragma unroll
        for (int v = 0; v < 8; ++v) {
            int m = v + 8 * half;
            float sv = s_lds[head_w * 16 + m];
            float inv = (sv > 0.f) ? (1.f / sv) : 0.f;
            out[(size_t)(i0 + m) * HF + col] = acc[q][v] * inv;
        }
    }
}

// column-wise batchnorm (over N rows) + ELU; one block per column. Bias cancels under BN.
__global__ void bn_elu_k(const float* __restrict__ y, const float* __restrict__ gamma,
                         const float* __restrict__ beta, float* __restrict__ out,
                         int n, int c) {
    __shared__ float s1[256], s2[256];
    int col = blockIdx.x, t = threadIdx.x;
    float sum = 0.f, sq = 0.f;
    for (int i = t; i < n; i += 256) {
        float v = y[(size_t)i * c + col];
        sum += v; sq += v * v;
    }
    s1[t] = sum; s2[t] = sq;
    __syncthreads();
    for (int ofs = 128; ofs > 0; ofs >>= 1) {
        if (t < ofs) { s1[t] += s1[t + ofs]; s2[t] += s2[t + ofs]; }
        __syncthreads();
    }
    float mu   = s1[0] / n;
    float var  = s2[0] / n - mu * mu;
    float istd = rsqrtf(var + 1e-5f);
    float g = gamma[col], b = beta[col];
    for (int i = t; i < n; i += 256) {
        float v = (y[(size_t)i * c + col] - mu) * istd * g + b;
        out[(size_t)i * c + col] = (v > 0.f) ? v : (__expf(v) - 1.f);
    }
}

extern "C" void kernel_launch(void* const* d_in, const int* in_sizes, int n_in,
                              void* d_out, int out_size, void* d_ws, size_t ws_size,
                              hipStream_t stream) {
    const float* x   = (const float*)d_in[0];
    const int*   adj = (const int*)  d_in[1];
    const float* W1  = (const float*)d_in[2];
    const float* a1s = (const float*)d_in[3];
    const float* a1d = (const float*)d_in[4];
    const float* lw1 = (const float*)d_in[5];
    const float* g1  = (const float*)d_in[7];
    const float* be1 = (const float*)d_in[8];
    const float* W2  = (const float*)d_in[9];
    const float* a2s = (const float*)d_in[10];
    const float* a2d = (const float*)d_in[11];
    const float* lw2 = (const float*)d_in[12];
    const float* g2  = (const float*)d_in[14];
    const float* be2 = (const float*)d_in[15];
    float* outp = (float*)d_out;

    char* ws = (char*)d_ws;
    size_t o = 0;
    auto carve = [&](size_t bytes) { char* p = ws + o; o += (bytes + 255) & ~(size_t)255; return p; };

    unsigned* adjb = (unsigned*) carve((size_t)NN * NW * 4);
    _Float16* xh   = (_Float16*)carve((size_t)NN * INF_ * 2);
    _Float16* w1t  = (_Float16*)carve((size_t)INF_ * HF1 * 2);   // [HF1][INF_]
    float*    h1f  = (float*)   carve((size_t)NN * HF1 * 4);
    _Float16* h1t  = (_Float16*)carve((size_t)NN * HF1 * 2);     // [HF1][NN]
    float*    es1  = (float*)   carve((size_t)HH * NN * 4);
    float*    ed1  = (float*)   carve((size_t)HH * NN * 4);
    float*    x1f  = (float*)   carve((size_t)NN * HF1 * 4);
    _Float16* x1h  = (_Float16*)carve((size_t)NN * HF1 * 2);
    _Float16* lw1t = (_Float16*)carve((size_t)HF1 * HID1 * 2);   // [HID1][HF1]
    float*    y1f  = (float*)   carve((size_t)NN * HID1 * 4);
    float*    x2f  = (float*)   carve((size_t)NN * HID1 * 4);
    _Float16* x2h  = (_Float16*)carve((size_t)NN * HID1 * 2);
    _Float16* w2t  = (_Float16*)carve((size_t)HID1 * HF2 * 2);   // [HF2][HID1]
    float*    h2f  = (float*)   carve((size_t)NN * HF2 * 4);
    _Float16* h2t  = (_Float16*)carve((size_t)NN * HF2 * 2);     // [HF2][NN]
    float*    es2  = (float*)   carve((size_t)HH * NN * 4);
    float*    ed2  = (float*)   carve((size_t)HH * NN * 4);
    float*    x3f  = (float*)   carve((size_t)NN * HF2 * 4);
    _Float16* x3h  = (_Float16*)carve((size_t)NN * HF2 * 2);
    _Float16* lw2t = (_Float16*)carve((size_t)HF2 * OUTF * 2);   // [OUTF][HF2]
    float*    y2f  = (float*)   carve((size_t)NN * OUTF * 4);

    auto cvt = [&](const float* src, _Float16* dst, int n) {
        f32_to_f16_k<<<(n + 255) / 256, 256, 0, stream>>>(src, dst, n);
    };
    auto cvtT = [&](const float* src, _Float16* dst, int R, int C) {
        f32_to_f16_tr_k<<<dim3(C / 16, R / 16), 256, 0, stream>>>(src, dst, R, C);
    };

    // ---- one-time packs ----
    adj_to_bits_k<<<(NN * NW + 255) / 256, 256, 0, stream>>>(adj, adjb);

    // ---- Layer 1 GAT ----
    cvt(x, xh, NN * INF_);
    cvtT(W1, w1t, INF_, HF1);
    gemm_f16_wmma<<<dim3(HF1 / 16, NN / 16), 32, 0, stream>>>(xh, w1t, h1f, NN, HF1, INF_);
    cvtT(h1f, h1t, NN, HF1);
    attn_scores_k<<<(NN * HH + 255) / 256, 256, 0, stream>>>(h1f, a1s, a1d, es1, ed1, NN, F1);
    gat_aggregate<4, F1><<<NN / 16, 256, 0, stream>>>(h1t, adjb, es1, ed1, x1f);

    // ---- Linear + BN + ELU ----
    cvt(x1f, x1h, NN * HF1);
    cvtT(lw1, lw1t, HF1, HID1);
    gemm_f16_wmma<<<dim3(HID1 / 16, NN / 16), 32, 0, stream>>>(x1h, lw1t, y1f, NN, HID1, HF1);
    bn_elu_k<<<HID1, 256, 0, stream>>>(y1f, g1, be1, x2f, NN, HID1);

    // ---- Layer 2 GAT ----
    cvt(x2f, x2h, NN * HID1);
    cvtT(W2, w2t, HID1, HF2);
    gemm_f16_wmma<<<dim3(HF2 / 16, NN / 16), 32, 0, stream>>>(x2h, w2t, h2f, NN, HF2, HID1);
    cvtT(h2f, h2t, NN, HF2);
    attn_scores_k<<<(NN * HH + 255) / 256, 256, 0, stream>>>(h2f, a2s, a2d, es2, ed2, NN, F2);
    gat_aggregate<1, F2><<<NN / 16, 256, 0, stream>>>(h2t, adjb, es2, ed2, x3f);

    // ---- Final linear + BN + ELU ----
    cvt(x3f, x3h, NN * HF2);
    cvtT(lw2, lw2t, HF2, OUTF);
    gemm_f16_wmma<<<dim3(OUTF / 16, NN / 16), 32, 0, stream>>>(x3h, lw2t, y2f, NN, OUTF, HF2);
    bn_elu_k<<<OUTF, 256, 0, stream>>>(y2f, g2, be2, outp, NN, OUTF);
}
